// RotEncoder_78417512891252
// MI455X (gfx1250) — matrix operand order repaired and checked by reading, not compiled, not run on hardware
//
#include <hip/hip_runtime.h>
#include <math.h>
#include <stdint.h>

// ---------------------------------------------------------------------------
// Problem constants (from the reference)
// ---------------------------------------------------------------------------
#define B_    4
#define N_    4096
#define M_    64
#define K_    3
#define R_    4
#define FEAT_ 1024
#define KN_   (K_ * N_)          // 12288
#define BR_   (B_ * R_)          // 16
#define COLS1_ ((long)BR_ * KN_) // 196608  (columns of first resnet GEMMs)
#define COLS2_ ((long)BR_ * M_)  // 1024    (columns of final resnet GEMMs)
#define EPS_BN 1e-5f

typedef __attribute__((ext_vector_type(16))) _Float16 v16h;
typedef __attribute__((ext_vector_type(8)))  _Float16 v8h;
typedef __attribute__((ext_vector_type(8)))  float    v8f;
typedef __attribute__((ext_vector_type(4)))  int      v4i;

// CDNA5 async global->LDS copy (ASYNCcnt-tracked). The builtin takes
// (v4i addrspace(1)* gsrc, v4i addrspace(3)* ldst, imm offset, imm cpol).
// Guarded so the build can never regress: if the builtin is absent we fall
// back to a synchronous register round-trip copy (LDS staging still gives the
// 4x B-operand reuse across the M-waves of the workgroup).
#if defined(__gfx1250__) && __has_builtin(__builtin_amdgcn_global_load_async_to_lds_b128)
#define HAS_ASYNC_LDS 1
#if __has_builtin(__builtin_amdgcn_s_wait_asynccnt)
#define WAIT_ASYNC(n) __builtin_amdgcn_s_wait_asynccnt(n)
#else
#define WAIT_ASYNC(n) asm volatile("s_wait_asynccnt " #n ::: "memory")
#endif
typedef __attribute__((address_space(1))) v4i as1_v4i;
typedef __attribute__((address_space(3))) v4i as3_v4i;
__device__ __forceinline__ void async_copy_b128(const _Float16* g, _Float16* l) {
  // global flat address == AS1 address; shared flat address low 32 bits ==
  // wave-relative LDS offset (ISA 10.2 aperture mapping)
  as1_v4i* gp = (as1_v4i*)(uintptr_t)g;
  as3_v4i* lp = (as3_v4i*)(uintptr_t)l;
  __builtin_amdgcn_global_load_async_to_lds_b128(gp, lp, 0, 0);
}
#endif

static __device__ __forceinline__ void rot_cs(int r, float& c, float& s) {
  // 2*pi*r/R with R=4  ->  r * pi/2
  float ang = 1.57079632679489662f * (float)r;
  c = cosf(ang);
  s = sinf(ang);
}

// ---------------------------------------------------------------------------
// misc small kernels
// ---------------------------------------------------------------------------
__global__ void zero_f32_kernel(float* __restrict__ p, long n) {
  long i = (long)blockIdx.x * blockDim.x + threadIdx.x;
  if (i < n) p[i] = 0.0f;
}

// pad/convert f32 weights [cout][cin] -> f16 [cout][cinp] (zero padded K)
__global__ void wconv_kernel(const float* __restrict__ W, _Float16* __restrict__ Wo,
                             int cout, int cin, int cinp) {
  long i = (long)blockIdx.x * blockDim.x + threadIdx.x;
  long total = (long)cout * cinp;
  if (i >= total) return;
  int co = (int)(i / cinp);
  int k  = (int)(i - (long)co * cinp);
  Wo[i] = (k < cin) ? (_Float16)W[(long)co * cin + k] : (_Float16)0.0f;
}

// ---------------------------------------------------------------------------
// top-3 nearest nodes per point; also accumulate per-node counts and coord sums
// ---------------------------------------------------------------------------
__global__ void topk_kernel(const float* __restrict__ x, const float* __restrict__ node,
                            int* __restrict__ min_idx, float* __restrict__ counts,
                            float* __restrict__ sums) {
  int i = blockIdx.x * blockDim.x + threadIdx.x;
  if (i >= B_ * N_) return;
  int b = i / N_, n = i - b * N_;
  float x0 = x[((long)b * 3 + 0) * N_ + n];
  float x1 = x[((long)b * 3 + 1) * N_ + n];
  float x2 = x[((long)b * 3 + 2) * N_ + n];
  float d0 = 3.4e38f, d1 = 3.4e38f, d2 = 3.4e38f;
  int i0 = 0, i1 = 0, i2 = 0;
  for (int m = 0; m < M_; ++m) {
    float e0 = x0 - node[((long)b * 3 + 0) * M_ + m];
    float e1 = x1 - node[((long)b * 3 + 1) * M_ + m];
    float e2 = x2 - node[((long)b * 3 + 2) * M_ + m];
    float d = e0 * e0 + e1 * e1 + e2 * e2;
    if (d < d0)      { d2 = d1; i2 = i1; d1 = d0; i1 = i0; d0 = d; i0 = m; }
    else if (d < d1) { d2 = d1; i2 = i1; d1 = d;  i1 = m; }
    else if (d < d2) { d2 = d;  i2 = m; }
  }
  // min_idx[b, kk*N + n] = idx[b,n,kk]
  min_idx[(long)b * KN_ + 0 * N_ + n] = i0;
  min_idx[(long)b * KN_ + 1 * N_ + n] = i1;
  min_idx[(long)b * KN_ + 2 * N_ + n] = i2;
  int id[3] = {i0, i1, i2};
#pragma unroll
  for (int kk = 0; kk < 3; ++kk) {
    int m = id[kk];
    atomicAdd(&counts[b * M_ + m], 1.0f);
    atomicAdd(&sums[((long)b * 3 + 0) * M_ + m], x0);
    atomicAdd(&sums[((long)b * 3 + 1) * M_ + m], x1);
    atomicAdd(&sums[((long)b * 3 + 2) * M_ + m], x2);
  }
}

// node_rot[b][r][3][M] = rot_r * (sums/(counts+1e-5))
__global__ void node_rot_kernel(const float* __restrict__ counts, const float* __restrict__ sums,
                                float* __restrict__ node_rot) {
  int i = blockIdx.x * blockDim.x + threadIdx.x;
  if (i >= B_ * R_ * M_) return;
  int m = i % M_;
  int r = (i / M_) % R_;
  int b = i / (M_ * R_);
  float cnt = counts[b * M_ + m] + 1e-5f;
  float c0 = sums[((long)b * 3 + 0) * M_ + m] / cnt;
  float c1 = sums[((long)b * 3 + 1) * M_ + m] / cnt;
  float c2 = sums[((long)b * 3 + 2) * M_ + m] / cnt;
  float c, s; rot_cs(r, c, s);
  long base = ((long)(b * R_ + r) * 3) * M_ + m;
  node_rot[base + 0 * M_] = c * c0 - s * c1;
  node_rot[base + 1 * M_] = s * c0 + c * c1;
  node_rot[base + 2 * M_] = c2;
}

// Build first-resnet input X0[col][32] f16: rows 0..2 x_dec, 3..5 sn_rot, 6..31 zero pad
__global__ void build_x0_kernel(const float* __restrict__ x, const float* __restrict__ sn,
                                const int* __restrict__ min_idx, const float* __restrict__ node_rot,
                                _Float16* __restrict__ X0) {
  long col = (long)blockIdx.x * blockDim.x + threadIdx.x;
  if (col >= COLS1_) return;
  int br = (int)(col / KN_);
  int j  = (int)(col - (long)br * KN_);
  int b = br >> 2, r = br & 3;
  int n = j % N_;
  float x0 = x[((long)b * 3 + 0) * N_ + n], x1 = x[((long)b * 3 + 1) * N_ + n], x2 = x[((long)b * 3 + 2) * N_ + n];
  float s0 = sn[((long)b * 3 + 0) * N_ + n], s1 = sn[((long)b * 3 + 1) * N_ + n], s2 = sn[((long)b * 3 + 2) * N_ + n];
  float c, s; rot_cs(r, c, s);
  float xr0 = c * x0 - s * x1, xr1 = s * x0 + c * x1, xr2 = x2;
  float sr0 = c * s0 - s * s1, sr1 = s * s0 + c * s1, sr2 = s2;
  int m = min_idx[(long)b * KN_ + j];
  const float* nr = node_rot + ((long)(b * R_ + r) * 3) * M_;
  _Float16* d = X0 + col * 32;
  d[0] = (_Float16)(xr0 - nr[0 * M_ + m]);
  d[1] = (_Float16)(xr1 - nr[1 * M_ + m]);
  d[2] = (_Float16)(xr2 - nr[2 * M_ + m]);
  d[3] = (_Float16)sr0; d[4] = (_Float16)sr1; d[5] = (_Float16)sr2;
#pragma unroll
  for (int q = 6; q < 32; ++q) d[q] = (_Float16)0.0f;
}

// ---------------------------------------------------------------------------
// WMMA GEMM + BN-statistics kernel, LDS-staged B with async double buffering.
//   Y[cols][cout] (f16 pre-BN) = W[cout][Kdim] * X[cols][ldx] (rows K-contig.)
//   stats[ch*2+{0,1}] += sum(y), sum(y^2)
// Workgroup = 8 waves, tile 64(M) x 128(N); each wave: 16(M) x 64(N).
// B tile (128 cols x 32 K, f16) is staged in LDS once per workgroup (4x reuse
// across the M-waves) via global_load_async_to_lds_b128, double buffered so
// copies of slab i+1 overlap WMMAs of slab i (handoff: s_wait_asynccnt 2 +
// workgroup barrier). LDS column stride padded to 80B -> conflict-free reads.
// Operand layouts per CDNA5 ISA 7.12.2 (wave32):
//   A 16x32 f16 : lane<16 row M=lane,     K {k0..k0+7, k0+16..k0+23}
//                 lane>=16 row M=lane-16, K {k0+8..k0+15, k0+24..k0+31}
//   B 32x16 f16 : lane<16 col N=lane,     K {k0..k0+7, k0+16..k0+23}
//                 lane>=16 col N=lane-16, K {k0+8..k0+15, k0+24..k0+31}
//   C/D 16x16 f32: VGPR v -> M = v + 8*(lane>=16), N = lane&15
// ---------------------------------------------------------------------------
#define BSTR 40  // halfs per LDS column slot (80 B: 64 B data + 16 B pad)

__global__ __launch_bounds__(256)
void gemm_bn_wmma_kernel(const _Float16* __restrict__ Wf, const _Float16* __restrict__ Xt,
                         _Float16* __restrict__ Yt, float* __restrict__ stats,
                         int Kdim, int ldx, int cout) {
  __shared__ _Float16 smem[2 * 128 * BSTR];  // 20 KB, double buffered

  const int tid  = threadIdx.x;
  const int lane = tid & 31;
  const int wv   = tid >> 5;
  const int wm   = wv & 3;           // 4 waves along M
  const int wn   = wv >> 2;          // 2 waves along N
  const int hs   = lane >> 4;        // half-wave select
  const int lq   = lane & 15;
  const int  m_base = blockIdx.y * 64 + wm * 16;
  const long n_blk  = (long)blockIdx.x * 128;

  // copy mapping: each thread moves two 16B chunks of the 128x32 B tile
  const int cp_col  = tid >> 1;            // 0..127
  const int cp_part = (tid & 1) * 2;       // {0,2} -> 16B chunks {0,1} or {2,3}
  const _Float16* gsrc = Xt + (n_blk + cp_col) * (long)ldx + cp_part * 8;
  _Float16* ldst = &smem[cp_col * BSTR + cp_part * 8];

  v8f acc[4];
#pragma unroll
  for (int t = 0; t < 4; ++t)
#pragma unroll
    for (int v = 0; v < 8; ++v) acc[t][v] = 0.0f;

  const _Float16* wrow = Wf + (long)(m_base + lq) * Kdim;
  const int T = Kdim >> 5;  // K-slabs of 32

  // prime buffer 0
  {
    const _Float16* g = gsrc;
    _Float16* l = ldst;
#ifdef HAS_ASYNC_LDS
    async_copy_b128(g, l);
    async_copy_b128(g + 8, l + 8);
#else
    *(v4i*)l       = *(const v4i*)g;
    *(v4i*)(l + 8) = *(const v4i*)(g + 8);
#endif
  }

  for (int i = 0; i < T; ++i) {
    if (i + 1 < T) {  // launch copies of the next K-slab into the other buffer
      const _Float16* g = gsrc + ((i + 1) << 5);
      _Float16* l = ldst + ((i + 1) & 1) * (128 * BSTR);
#ifdef HAS_ASYNC_LDS
      async_copy_b128(g, l);
      async_copy_b128(g + 8, l + 8);
      WAIT_ASYNC(2);  // slab i's two async ops (issued before) are complete
#else
      *(v4i*)l       = *(const v4i*)g;
      *(v4i*)(l + 8) = *(const v4i*)(g + 8);
#endif
    } else {
#ifdef HAS_ASYNC_LDS
      WAIT_ASYNC(0);
#endif
    }
    __syncthreads();  // slab i visible to all waves

    const int k0 = i << 5;
    // A fragment straight from global (W is small & L2-resident)
    v8h alo = *(const v8h*)(wrow + k0 + 8 * hs);
    v8h ahi = *(const v8h*)(wrow + k0 + 16 + 8 * hs);
    v16h a;
#pragma unroll
    for (int q = 0; q < 8; ++q) { a[q] = alo[q]; a[q + 8] = ahi[q]; }

    const _Float16* lbase = &smem[(i & 1) * (128 * BSTR)];
#pragma unroll
    for (int t = 0; t < 4; ++t) {
      const _Float16* lb = lbase + (wn * 64 + t * 16 + lq) * BSTR + hs * 8;
      v8h blo = *(const v8h*)lb;
      v8h bhi = *(const v8h*)(lb + 16);
      v16h bf;
#pragma unroll
      for (int q = 0; q < 8; ++q) { bf[q] = blo[q]; bf[q + 8] = bhi[q]; }
      acc[t] = __builtin_amdgcn_wmma_f32_16x16x32_f16(
          /*neg_a=*/false, a, /*neg_b=*/false, bf,
          /*c_mod=*/(short)0, acc[t], /*reuse_a=*/false, /*reuse_b=*/false);
    }
    __syncthreads();  // everyone done reading slab i before its buffer is refilled
  }

  // epilogue: write f16 pre-BN output + per-channel sum / sumsq
  float s[8], sq[8];
#pragma unroll
  for (int v = 0; v < 8; ++v) { s[v] = 0.0f; sq[v] = 0.0f; }
#pragma unroll
  for (int t = 0; t < 4; ++t) {
    const long col = n_blk + wn * 64 + t * 16 + lq;
    v8h o;
#pragma unroll
    for (int v = 0; v < 8; ++v) {
      float y = acc[t][v];
      o[v] = (_Float16)y;
      s[v] += y;
      sq[v] += y * y;
    }
    *(v8h*)(Yt + col * (long)cout + m_base + 8 * hs) = o;  // 8 consecutive channels
  }
#pragma unroll
  for (int off = 8; off >= 1; off >>= 1) {
#pragma unroll
    for (int v = 0; v < 8; ++v) {
      s[v]  += __shfl_xor(s[v],  off, 32);
      sq[v] += __shfl_xor(sq[v], off, 32);
    }
  }
  if (lq == 0) {
#pragma unroll
    for (int v = 0; v < 8; ++v) {
      const int ch = m_base + 8 * hs + v;
      atomicAdd(&stats[2 * ch + 0], s[v]);
      atomicAdd(&stats[2 * ch + 1], sq[v]);
    }
  }
}

// per-channel scale/shift from stats (conv bias cancels under train-mode BN)
__global__ void bn_finalize_kernel(const float* __restrict__ stats, const float* __restrict__ g,
                                   const float* __restrict__ be, float* __restrict__ sshift,
                                   int cout, float inv_n) {
  int i = blockIdx.x * blockDim.x + threadIdx.x;
  if (i >= cout) return;
  float mu  = stats[2 * i + 0] * inv_n;
  float var = stats[2 * i + 1] * inv_n - mu * mu;
  float sc  = g[i] * rsqrtf(var + EPS_BN);
  sshift[2 * i + 0] = sc;
  sshift[2 * i + 1] = be[i] - mu * sc;
}

// act = relu(y*scale+shift); write into concat destination rows [off, off+cout)
__global__ void bn_apply_kernel(const _Float16* __restrict__ Yt, const float* __restrict__ sshift,
                                _Float16* __restrict__ dst, int cout, int ldd, int off, long total8) {
  long i = (long)blockIdx.x * blockDim.x + threadIdx.x;
  if (i >= total8) return;
  int  cg  = cout >> 3;
  long col = i / cg;
  int  c   = (int)(i - col * cg) * 8;
  v8h y = *(const v8h*)(Yt + col * (long)cout + c);
  v8h o;
#pragma unroll
  for (int v = 0; v < 8; ++v) {
    float val = (float)y[v] * sshift[2 * (c + v)] + sshift[2 * (c + v) + 1];
    o[v] = (_Float16)fmaxf(val, 0.0f);
  }
  *(v8h*)(dst + col * (long)ldd + off + c) = o;
}

// layer-4 apply fused with per-node segment-max (post-ReLU >= 0 -> int-punned atomicMax)
__global__ void bn_apply_segmax_kernel(const _Float16* __restrict__ Yt, const float* __restrict__ sshift,
                                       const int* __restrict__ min_idx, float* __restrict__ f1max,
                                       float* __restrict__ f1col0, long total8) {
  long i = (long)blockIdx.x * blockDim.x + threadIdx.x;
  if (i >= total8) return;
  const int cout = 384, cg = 384 / 8;
  long col = i / cg;
  int  c   = (int)(i - col * cg) * 8;
  int br = (int)(col / KN_);
  int j  = (int)(col - (long)br * KN_);
  int b  = br >> 2;
  int m  = min_idx[(long)b * KN_ + j];
  v8h y = *(const v8h*)(Yt + col * (long)cout + c);
#pragma unroll
  for (int v = 0; v < 8; ++v) {
    float val = fmaxf((float)y[v] * sshift[2 * (c + v)] + sshift[2 * (c + v) + 1], 0.0f);
    atomicMax((int*)&f1max[((long)br * 384 + c + v) * M_ + m], __float_as_int(val));
    if (j == 0) f1col0[br * 384 + c + v] = val;  // reference fallback f1[:, :, 0:1]
  }
}

// empty-node fixup + build final-resnet input Xf0[col2][416] (3 node_rot + 384 f1_max + pad)
__global__ void build_xf_kernel(const float* __restrict__ counts, const float* __restrict__ node_rot,
                                const float* __restrict__ f1max, const float* __restrict__ f1col0,
                                _Float16* __restrict__ Xf0) {
  int col2 = blockIdx.x * blockDim.x + threadIdx.x;
  if (col2 >= (int)COLS2_) return;
  int br = col2 / M_, m = col2 % M_;
  int b = br >> 2, r = br & 3;
  bool empty = counts[b * M_ + m] <= 0.0f;
  _Float16* d = Xf0 + (long)col2 * 416;
  long nrb = ((long)(b * R_ + r) * 3) * M_ + m;
  d[0] = (_Float16)node_rot[nrb + 0 * M_];
  d[1] = (_Float16)node_rot[nrb + 1 * M_];
  d[2] = (_Float16)node_rot[nrb + 2 * M_];
  for (int c = 0; c < 384; ++c) {
    float v = empty ? f1col0[br * 384 + c] : f1max[((long)br * 384 + c) * M_ + m];
    d[3 + c] = (_Float16)v;
  }
#pragma unroll
  for (int q = 387; q < 416; ++q) d[q] = (_Float16)0.0f;
}

// out[b][f] = max over (r,m) of final activation
__global__ void out_max_kernel(const _Float16* __restrict__ AfOut, float* __restrict__ out) {
  int i = blockIdx.x * blockDim.x + threadIdx.x;
  if (i >= B_ * FEAT_) return;
  int b = i / FEAT_, f = i - b * FEAT_;
  float mx = -3.4e38f;
  for (int rm = 0; rm < R_ * M_; ++rm) {
    float v = (float)AfOut[((long)(b * R_ * M_ + rm)) * 1024 + f];
    mx = fmaxf(mx, v);
  }
  out[i] = mx;
}

// ---------------------------------------------------------------------------
// host orchestration
// ---------------------------------------------------------------------------
static inline long cdiv(long a, long b) { return (a + b - 1) / b; }

extern "C" void kernel_launch(void* const* d_in, const int* in_sizes, int n_in,
                              void* d_out, int out_size, void* d_ws, size_t ws_size,
                              hipStream_t stream) {
  const float* x    = (const float*)d_in[0];
  const float* sn   = (const float*)d_in[1];
  const float* node = (const float*)d_in[2];
  // d_in[3] = node_knn_I (unused)
  // first_params: layers at indices 4,8,12,16 ; final_params: 20,24,28,32 ; each (W,b,g,be)

  const int widx[8]  = {4, 8, 12, 16, 20, 24, 28, 32};
  const int couts[8] = {64, 128, 256, 384, 512, 512, 768, 1024};
  const int cins[8]  = {6, 64, 128, 320, 387, 512, 512, 1280};
  const int cinp[8]  = {32, 64, 128, 320, 416, 512, 512, 1280};

  // ---- workspace carve-up (256B aligned) ----
  char* p = (char*)d_ws;
  auto alloc = [&](size_t bytes) -> void* {
    void* r = (void*)p;
    p += (bytes + 255) & ~(size_t)255;
    return r;
  };
  int*   min_idx  = (int*)  alloc((size_t)B_ * KN_ * 4);
  float* counts   = (float*)alloc((size_t)B_ * M_ * 4);
  float* sums     = (float*)alloc((size_t)B_ * 3 * M_ * 4);
  float* node_rot = (float*)alloc((size_t)B_ * R_ * 3 * M_ * 4);
  float* stats    = (float*)alloc((size_t)1024 * 2 * 4);
  float* sshift   = (float*)alloc((size_t)1024 * 2 * 4);
  _Float16* X0    = (_Float16*)alloc((size_t)COLS1_ * 32 * 2);
  _Float16* Ybuf  = (_Float16*)alloc((size_t)COLS1_ * 384 * 2);  // reused pre-BN buffer
  _Float16* Acat  = (_Float16*)alloc((size_t)COLS1_ * 320 * 2);  // [h0(64) | h2(256)]
  _Float16* A1    = (_Float16*)alloc((size_t)COLS1_ * 128 * 2);
  float* f1max    = (float*)alloc((size_t)BR_ * 384 * M_ * 4);
  float* f1col0   = (float*)alloc((size_t)BR_ * 384 * 4);
  _Float16* Xf0   = (_Float16*)alloc((size_t)COLS2_ * 416 * 2);
  _Float16* XfCat = (_Float16*)alloc((size_t)COLS2_ * 1280 * 2); // [hf0(512) | hf2(768)]
  _Float16* XfB   = (_Float16*)alloc((size_t)COLS2_ * 512 * 2);
  _Float16* AfOut = (_Float16*)alloc((size_t)COLS2_ * 1024 * 2);
  _Float16* Wb[8];
  for (int l = 0; l < 8; ++l) Wb[l] = (_Float16*)alloc((size_t)couts[l] * cinp[l] * 2);

  auto zero = [&](float* ptr, long n) {
    zero_f32_kernel<<<(unsigned)cdiv(n, 256), 256, 0, stream>>>(ptr, n);
  };

  // ---- weight conversion (f32 -> f16, K zero-padded) ----
  for (int l = 0; l < 8; ++l) {
    long total = (long)couts[l] * cinp[l];
    wconv_kernel<<<(unsigned)cdiv(total, 256), 256, 0, stream>>>(
        (const float*)d_in[widx[l]], Wb[l], couts[l], cins[l], cinp[l]);
  }

  // ---- assignment / geometry ----
  zero(counts, (long)B_ * M_);
  zero(sums, (long)B_ * 3 * M_);
  topk_kernel<<<(unsigned)cdiv((long)B_ * N_, 256), 256, 0, stream>>>(x, node, min_idx, counts, sums);
  node_rot_kernel<<<(unsigned)cdiv((long)B_ * R_ * M_, 256), 256, 0, stream>>>(counts, sums, node_rot);
  build_x0_kernel<<<(unsigned)cdiv(COLS1_, 256), 256, 0, stream>>>(x, sn, min_idx, node_rot, X0);

  const float invn1 = 1.0f / (float)COLS1_;
  const float invn2 = 1.0f / (float)COLS2_;

  auto gemm_layer = [&](int l, const _Float16* Xin, int ldx, long cols, float invn) {
    zero(stats, 2L * couts[l]);
    dim3 grid((unsigned)(cols / 128), (unsigned)(couts[l] / 64));
    gemm_bn_wmma_kernel<<<grid, 256, 0, stream>>>(Wb[l], Xin, Ybuf, stats, cinp[l], ldx, couts[l]);
    bn_finalize_kernel<<<(unsigned)cdiv(couts[l], 256), 256, 0, stream>>>(
        stats, (const float*)d_in[widx[l] + 2], (const float*)d_in[widx[l] + 3],
        sshift, couts[l], invn);
  };
  auto apply = [&](int cout, _Float16* dst, int ldd, int off, long cols) {
    long total8 = cols * (cout / 8);
    bn_apply_kernel<<<(unsigned)cdiv(total8, 256), 256, 0, stream>>>(Ybuf, sshift, dst, cout, ldd, off, total8);
  };

  // ---- PointResNet #1 : 196608 columns ----
  gemm_layer(0, X0, 32, COLS1_, invn1);    apply(64,  Acat, 320, 0,  COLS1_);  // h0 -> Acat[0:64)
  gemm_layer(1, Acat, 320, COLS1_, invn1); apply(128, A1,   128, 0,  COLS1_);  // h1
  gemm_layer(2, A1, 128, COLS1_, invn1);   apply(256, Acat, 320, 64, COLS1_);  // h2 -> Acat[64:320)
  // layer 4 (320 -> 384) fused with segment-max
  zero(f1max, (long)BR_ * 384 * M_);
  zero(f1col0, (long)BR_ * 384);
  gemm_layer(3, Acat, 320, COLS1_, invn1);
  {
    long total8 = COLS1_ * (384 / 8);
    bn_apply_segmax_kernel<<<(unsigned)cdiv(total8, 256), 256, 0, stream>>>(
        Ybuf, sshift, min_idx, f1max, f1col0, total8);
  }
  build_xf_kernel<<<(unsigned)cdiv(COLS2_, 256), 256, 0, stream>>>(counts, node_rot, f1max, f1col0, Xf0);

  // ---- PointResNet #2 : 1024 columns ----
  gemm_layer(4, Xf0, 416, COLS2_, invn2);    apply(512,  XfCat, 1280, 0,   COLS2_); // hf0
  gemm_layer(5, XfCat, 1280, COLS2_, invn2); apply(512,  XfB,   512,  0,   COLS2_); // hf1
  gemm_layer(6, XfB, 512, COLS2_, invn2);    apply(768,  XfCat, 1280, 512, COLS2_); // hf2
  gemm_layer(7, XfCat, 1280, COLS2_, invn2); apply(1024, AfOut, 1024, 0,   COLS2_); // final feat

  out_max_kernel<<<(unsigned)cdiv((long)B_ * FEAT_, 256), 256, 0, stream>>>(AfOut, (float*)d_out);
}
